// BILSTM_CRF_85280870629978
// MI455X (gfx1250) — matrix operand order repaired and checked by reading, not compiled, not run on hardware
//
#include <hip/hip_runtime.h>
#include <math.h>

// ---------------------------------------------------------------------------
// BiLSTM-CRF forward loss for MI455X (gfx1250, wave32, WMMA).
// B=64, T=512, V=30000, E=256, H=512, C=9.
//   1) embed_gather : X[T][B][E] bf16  (time-major, feeds LSTM A-matrix)
//   2) pack_weights : Wcat[dir][4H][E+H] bf16  (fused input+recurrent GEMM)
//   3) bilstm       : 2 workgroups (one per direction), 32 waves each.
//                     Per step: G = [x_t | h] @ Wcat^T + b via
//                     v_wmma_f32_16x16x32_bf16 (K=768 -> 24 WMMA per tile,
//                     4 gate tiles x 4 row tiles per wave = 384 WMMA/step).
//                     A-matrix lives in LDS (64x776 bf16 ~ 97KB; CDNA5 has
//                     320KB/WGP). c-state lives in VGPRs across all steps.
//                     x_t staged via GLOBAL_LOAD_ASYNC_TO_LDS_B128 (ASYNCcnt).
//   4) fc_emis      : out @ Wfc^T + log_softmax  (VALU, tiny)
//   5) crf          : gold score + forward algorithm, 1 wave per sequence
//   6) reduce       : sum over batch -> d_out[0]
// ---------------------------------------------------------------------------

typedef __bf16 bf16;
typedef __attribute__((ext_vector_type(16))) __bf16 v16bf;
typedef __attribute__((ext_vector_type(8)))  float  v8f;

static constexpr int Bn   = 64;
static constexpr int Tn   = 512;
static constexpr int En   = 256;
static constexpr int Hn   = 512;
static constexpr int G4   = 4 * Hn;        // 2048 gate columns
static constexpr int Kt   = En + Hn;       // 768 fused K
static constexpr int Cn   = 9;
static constexpr int LDSTR = Kt + 8;       // 776 bf16 per LDS row (pad)

union FragBF { v16bf v; uint4 u[2]; };
union Acc8   { v8f   v; float f[8]; };

#if __has_builtin(__builtin_amdgcn_global_load_async_to_lds_b128)
#define HAS_ASYNC_LDS 1
typedef int v4i_vec __attribute__((__vector_size__(16)));
typedef __attribute__((address_space(1))) v4i_vec gv4i;   // global int4
typedef __attribute__((address_space(3))) v4i_vec lv4i;   // LDS int4
#else
#define HAS_ASYNC_LDS 0
#endif

// v_rcp_f32 instead of IEEE div_scale/div_fmas chains (critical path is VALU)
__device__ __forceinline__ float fast_rcp(float x) {
  return __builtin_amdgcn_rcpf(x);
}
__device__ __forceinline__ float fast_sigmoid(float x) {
  return fast_rcp(1.f + __expf(-x));
}
// branch-free tanh on v_exp_f32 (avoids libm's exec-mask range-check branches)
__device__ __forceinline__ float fast_tanh(float x) {
  const float e = __expf(-2.f * fabsf(x));
  return copysignf((1.f - e) * fast_rcp(1.f + e), x);
}

// ------------------------------- kernels -----------------------------------

__global__ __launch_bounds__(256) void embed_gather_kernel(
    const int* __restrict__ x, const float* __restrict__ emb,
    bf16* __restrict__ X)
{
  const int idx = blockIdx.x * 256 + threadIdx.x;   // (t, b, e/4)
  if (idx >= Tn * Bn * (En / 4)) return;
  const int e4 = idx & 63;
  const int tb = idx >> 6;
  const int b  = tb & (Bn - 1);
  const int t  = tb >> 6;
  const int tok = x[b * Tn + t];
  const float4 v = ((const float4*)(emb + (size_t)tok * En))[e4];
  union { unsigned long long q; bf16 e[4]; } o;
  o.e[0] = (bf16)v.x; o.e[1] = (bf16)v.y; o.e[2] = (bf16)v.z; o.e[3] = (bf16)v.w;
  *(unsigned long long*)(X + ((size_t)t * Bn + b) * En + e4 * 4) = o.q;
}

__global__ __launch_bounds__(256) void pack_weights_kernel(
    const float* __restrict__ Wih_f, const float* __restrict__ Whh_f,
    const float* __restrict__ Wih_b, const float* __restrict__ Whh_b,
    bf16* __restrict__ Wcat)
{
  const int idx = blockIdx.x * 256 + threadIdx.x;
  if (idx >= 2 * G4 * Kt) return;
  const int k = idx % Kt;
  const int g = (idx / Kt) % G4;
  const int d = idx / (Kt * G4);
  const float* Wih = d ? Wih_b : Wih_f;
  const float* Whh = d ? Whh_b : Whh_f;
  const float v = (k < En) ? Wih[(size_t)g * En + k]
                           : Whh[(size_t)g * Hn + (k - En)];
  Wcat[idx] = (bf16)v;
}

// One workgroup per direction. 32 waves; wave w owns h-columns [16w, 16w+16).
__global__ __launch_bounds__(1024) void bilstm_kernel(
    const bf16* __restrict__ X,        // [T][B][E]
    const bf16* __restrict__ Wcat,     // [2][G4][Kt]
    const float* __restrict__ bias_f, const float* __restrict__ bias_b,
    const int* __restrict__ seq_len,
    bf16* __restrict__ hout)           // [B][T][2H]
{
  extern __shared__ char smem[];
  bf16* shA  = (bf16*)smem;                               // [64][LDSTR]
  unsigned short* shU = (unsigned short*)smem;            // bf16 raw-bit view
  int* sh_len = (int*)(smem + (size_t)Bn * LDSTR * sizeof(bf16));

  const int dir   = blockIdx.x;
  const bf16* W   = Wcat + (size_t)dir * G4 * Kt;
  const float* bias = dir ? bias_b : bias_f;

  const int tid  = threadIdx.x;
  const int wave = tid >> 5;
  const int lane = tid & 31;
  const int nn   = lane & 15;   // N / row-in-tile index
  const int half = lane >> 4;
  const int j0   = wave * 16;   // h-column tile base

  if (tid < Bn) sh_len[tid] = seq_len[tid];
  for (int i = tid; i < Bn * LDSTR; i += 1024) shA[i] = (bf16)0.f;

  float bi[4];
  #pragma unroll
  for (int q = 0; q < 4; ++q) bi[q] = bias[q * Hn + j0 + nn];

  float cst[4][8];   // cell state, lives in VGPRs for all 512 steps
  #pragma unroll
  for (int mt = 0; mt < 4; ++mt)
    #pragma unroll
    for (int v = 0; v < 8; ++v) cst[mt][v] = 0.f;

  __syncthreads();

  // seq-lens packed 2-per-VGPR (rows b = mt*16 + {2k,2k+1} + 8*half)
  unsigned Lp[4][4];
  #pragma unroll
  for (int mt = 0; mt < 4; ++mt)
    #pragma unroll
    for (int k = 0; k < 4; ++k)
      Lp[mt][k] = (unsigned)sh_len[mt * 16 + 2 * k + 8 * half] |
                  ((unsigned)sh_len[mt * 16 + 2 * k + 1 + 8 * half] << 16);

  for (int step = 0; step < Tn; ++step) {
    const int t = dir ? (Tn - 1 - step) : step;

    // stage x_t into LDS columns [0, 256)
    {
      const uint4* src = (const uint4*)(X + (size_t)t * Bn * En);
      #pragma unroll
      for (int it = 0; it < 2; ++it) {
        const int i = tid + it * 1024;           // 2048 uint4 total
        const int r = i >> 5, cc = i & 31;
#if HAS_ASYNC_LDS
        __builtin_amdgcn_global_load_async_to_lds_b128(
            (gv4i*)(src + i), (lv4i*)(&shA[r * LDSTR + cc * 8]), 0, 0);
#else
        *(uint4*)(&shA[r * LDSTR + cc * 8]) = src[i];
#endif
      }
    }
    if (step + 1 < Tn) {  // prefetch next timestep (global_prefetch_b8)
      const int tnext = dir ? (Tn - 2 - step) : (step + 1);
      __builtin_prefetch((const char*)(X + (size_t)tnext * Bn * En) + tid * 32, 0, 0);
    }
#if HAS_ASYNC_LDS
#if __has_builtin(__builtin_amdgcn_s_wait_asynccnt)
    __builtin_amdgcn_s_wait_asynccnt(0);
#else
    asm volatile("s_wait_asynccnt 0" ::: "memory");
#endif
#endif
    __syncthreads();

    unsigned hmp[4][4];   // merged carried-h, raw bf16 bits packed 2-per-VGPR
    #pragma unroll 1
    for (int mt = 0; mt < 4; ++mt) {
      const int m0 = mt * 16;
      Acc8 acc[4];
      #pragma unroll
      for (int q = 0; q < 4; ++q)
        #pragma unroll
        for (int v = 0; v < 8; ++v) acc[q].f[v] = bi[q];

      const bf16* arow = &shA[(m0 + nn) * LDSTR];
      #pragma unroll 2
      for (int kc = 0; kc < Kt / 32; ++kc) {
        const int kb = kc * 32;
        // A frag: row m0+nn, K runs [half*8, +8) and [16+half*8, +8)
        FragBF fa;
        fa.u[0] = *(const uint4*)(arow + kb + half * 8);
        fa.u[1] = *(const uint4*)(arow + kb + 16 + half * 8);
        #pragma unroll
        for (int q = 0; q < 4; ++q) {
          // B frag: column (gate row of W), K runs [half*8, +8) and [16+half*8, +8)
          const bf16* brow = W + (size_t)(q * Hn + j0 + nn) * Kt + kb;
          FragBF fb;
          fb.u[0] = *(const uint4*)(brow + half * 8);
          fb.u[1] = *(const uint4*)(brow + 16 + half * 8);
          acc[q].v = __builtin_amdgcn_wmma_f32_16x16x32_bf16(
              false, fa.v, false, fb.v, (short)0, acc[q].v, false, false);
        }
      }

      // elementwise LSTM cell; torch gate order i, f, g, o  (branch-free)
      #pragma unroll
      for (int k = 0; k < 4; ++k) {
        unsigned short hb2[2];
        #pragma unroll
        for (int jj = 0; jj < 2; ++jj) {
          const int v = 2 * k + jj;
          const int b = m0 + v + 8 * half;
          const float ig = fast_sigmoid(acc[0].f[v]);
          const float fg = fast_sigmoid(acc[1].f[v]);
          const float gg = fast_tanh(acc[2].f[v]);
          const float og = fast_sigmoid(acc[3].f[v]);
          const float cn = fg * cst[mt][v] + ig * gg;
          const float hn = og * fast_tanh(cn);
          const int Lv = (int)((Lp[mt][k] >> (16 * jj)) & 0xffffu);
          const bool mvalid = (t < Lv);
          cst[mt][v] = mvalid ? cn : cst[mt][v];
          union { bf16 h; unsigned short u; } cv;
          cv.h = (bf16)hn;
          const unsigned short holdu = shU[(size_t)b * LDSTR + En + j0 + nn];
          hb2[jj] = mvalid ? cv.u : holdu;   // frozen carry (packed-seq)
          // scan output hs is the *unmasked* h_new (masked later in FC)
          hout[((size_t)b * Tn + t) * (2 * Hn) + dir * Hn + j0 + nn] = (bf16)hn;
        }
        hmp[mt][k] = (unsigned)hb2[0] | ((unsigned)hb2[1] << 16);
      }
    }
    __syncthreads();   // all WMMA reads of shA complete

    // write carried h back into LDS A-matrix columns [256, 768)
    #pragma unroll
    for (int mt = 0; mt < 4; ++mt)
      #pragma unroll
      for (int k = 0; k < 4; ++k) {
        const int b0 = mt * 16 + 2 * k + 8 * half;
        shU[(size_t)b0 * LDSTR + En + j0 + nn] = (unsigned short)hmp[mt][k];
        shU[(size_t)(b0 + 1) * LDSTR + En + j0 + nn] =
            (unsigned short)(hmp[mt][k] >> 16);
      }
    // next iteration's __syncthreads orders these writes before reads
  }
}

__global__ __launch_bounds__(256) void fc_emis_kernel(
    const bf16* __restrict__ hout, const float* __restrict__ Wfc,
    const int* __restrict__ seq_len, float* __restrict__ emis)
{
  __shared__ float sw[Cn * 2 * Hn];   // 36KB
  for (int i = threadIdx.x; i < Cn * 2 * Hn; i += 256) sw[i] = Wfc[i];
  __syncthreads();

  const int idx = blockIdx.x * 256 + threadIdx.x;   // b*T + t, exact grid
  const int b = idx >> 9, t = idx & 511;
  float logit[Cn];
  #pragma unroll
  for (int c = 0; c < Cn; ++c) logit[c] = 0.f;
  if (t < seq_len[b]) {   // masked rows: out=0 -> logits=0 (matches reference)
    const bf16* h = hout + (size_t)idx * (2 * Hn);
    for (int j = 0; j < 2 * Hn; j += 8) {
      union { uint4 u; bf16 e[8]; } hv;
      hv.u = *(const uint4*)(h + j);
      float hf[8];
      #pragma unroll
      for (int k = 0; k < 8; ++k) hf[k] = (float)hv.e[k];
      #pragma unroll
      for (int c = 0; c < Cn; ++c) {
        float s = logit[c];
        #pragma unroll
        for (int k = 0; k < 8; ++k) s += hf[k] * sw[c * (2 * Hn) + j + k];
        logit[c] = s;
      }
    }
  }
  float mx = logit[0];
  #pragma unroll
  for (int c = 1; c < Cn; ++c) mx = fmaxf(mx, logit[c]);
  float se = 0.f;
  #pragma unroll
  for (int c = 0; c < Cn; ++c) se += __expf(logit[c] - mx);
  const float lse = mx + __logf(se);
  #pragma unroll
  for (int c = 0; c < Cn; ++c) emis[(size_t)idx * Cn + c] = logit[c] - lse;
}

// One wave per sequence: gold-path score + forward-algorithm log-partition.
__global__ __launch_bounds__(32) void crf_kernel(
    const float* __restrict__ emis, const int* __restrict__ y,
    const int* __restrict__ seq_len, const float* __restrict__ start_t,
    const float* __restrict__ end_t, const float* __restrict__ trans,
    float* __restrict__ nll)
{
  const int b = blockIdx.x;
  const int lane = threadIdx.x;
  const int L = seq_len[b];
  const float* em = emis + (size_t)b * Tn * Cn;
  const int* yb = y + b * Tn;

  // numerator partials: emission + transition scores along gold path
  float part = 0.f;
  for (int t = lane; t < Tn; t += 32) {
    if (t < L) {
      part += em[t * Cn + yb[t]];
      if (t >= 1) part += trans[yb[t - 1] * Cn + yb[t]];
    }
  }
  #pragma unroll
  for (int off = 16; off > 0; off >>= 1) part += __shfl_down(part, off, 32);
  float num = 0.f;
  if (lane == 0) num = part + start_t[yb[0]] + end_t[yb[L - 1]];

  __shared__ float s_score[Cn];
  float trj[Cn];
  if (lane < Cn) {
    #pragma unroll
    for (int i = 0; i < Cn; ++i) trj[i] = trans[i * Cn + lane];
    s_score[lane] = start_t[lane] + em[lane];
  }
  __syncthreads();
  for (int t = 1; t < Tn; ++t) {
    float sc[Cn];
    #pragma unroll
    for (int i = 0; i < Cn; ++i) sc[i] = s_score[i];
    __syncthreads();
    if (lane < Cn && t < L) {
      float mx = -3.4e38f;
      #pragma unroll
      for (int i = 0; i < Cn; ++i) mx = fmaxf(mx, sc[i] + trj[i]);
      float se = 0.f;
      #pragma unroll
      for (int i = 0; i < Cn; ++i) se += __expf(sc[i] + trj[i] - mx);
      s_score[lane] = mx + __logf(se) + em[t * Cn + lane];
    }
    __syncthreads();
  }
  if (lane == 0) {
    float mx = -3.4e38f;
    #pragma unroll
    for (int j = 0; j < Cn; ++j) mx = fmaxf(mx, s_score[j] + end_t[j]);
    float se = 0.f;
    #pragma unroll
    for (int j = 0; j < Cn; ++j) se += __expf(s_score[j] + end_t[j] - mx);
    nll[b] = (mx + __logf(se)) - num;   // den - num
  }
}

__global__ __launch_bounds__(64) void reduce_kernel(
    const float* __restrict__ nll, float* __restrict__ out)
{
  __shared__ float s[Bn];
  const int tid = threadIdx.x;
  s[tid] = nll[tid];
  __syncthreads();
  #pragma unroll
  for (int off = 32; off > 0; off >>= 1) {
    if (tid < off) s[tid] += s[tid + off];
    __syncthreads();
  }
  if (tid == 0) out[0] = s[0];
}

// ------------------------------- launcher ----------------------------------

extern "C" void kernel_launch(void* const* d_in, const int* in_sizes, int n_in,
                              void* d_out, int out_size, void* d_ws, size_t ws_size,
                              hipStream_t stream) {
  (void)in_sizes; (void)n_in; (void)out_size; (void)ws_size;
  const int*   x       = (const int*)d_in[0];
  const int*   seq_len = (const int*)d_in[1];
  const int*   y       = (const int*)d_in[2];
  const float* emb     = (const float*)d_in[3];
  const float* Wih_f   = (const float*)d_in[4];
  const float* Whh_f   = (const float*)d_in[5];
  const float* b_f     = (const float*)d_in[6];
  const float* Wih_b   = (const float*)d_in[7];
  const float* Whh_b   = (const float*)d_in[8];
  const float* b_b     = (const float*)d_in[9];
  const float* Wfc     = (const float*)d_in[10];
  const float* start_t = (const float*)d_in[11];
  const float* end_t   = (const float*)d_in[12];
  const float* trans   = (const float*)d_in[13];
  float* out = (float*)d_out;

  char* ws = (char*)d_ws;
  // workspace layout (all 256B aligned)
  bf16*  X    = (bf16*)(ws);                         // 16,777,216 B
  bf16*  Wcat = (bf16*)(ws + 16777216);              //  6,291,456 B
  bf16*  hout = (bf16*)(ws + 23068672);              // 67,108,864 B
  float* emis = (float*)(ws + 90177536);             //  1,179,648 B
  float* nll  = (float*)(ws + 91357184);             //        256 B

  {
    const int total = Tn * Bn * (En / 4);
    embed_gather_kernel<<<(total + 255) / 256, 256, 0, stream>>>(x, emb, X);
  }
  {
    const int total = 2 * G4 * Kt;
    pack_weights_kernel<<<(total + 255) / 256, 256, 0, stream>>>(
        Wih_f, Whh_f, Wih_b, Whh_b, Wcat);
  }
  {
    const size_t smem = (size_t)Bn * LDSTR * sizeof(bf16) + Bn * sizeof(int);
    bilstm_kernel<<<2, 1024, smem, stream>>>(X, Wcat, b_f, b_b, seq_len, hout);
  }
  fc_emis_kernel<<<(Bn * Tn) / 256, 256, 0, stream>>>(hout, Wfc, seq_len, emis);
  crf_kernel<<<Bn, 32, 0, stream>>>(emis, y, seq_len, start_t, end_t, trans, nll);
  reduce_kernel<<<1, 64, 0, stream>>>(nll, out);
}
